// ReprogrammingLayer_44805098832441
// MI455X (gfx1250) — matrix operand order repaired and checked by reading, not compiled, not run on hardware
//
#include <hip/hip_runtime.h>
#include <hip/hip_bf16.h>

typedef __attribute__((ext_vector_type(16))) _Float16 v16h;
typedef __attribute__((ext_vector_type(8)))  _Float16 v8h;
typedef __attribute__((ext_vector_type(8)))  float    v8f;

#define BM      8
#define LQ      1024
#define VS      2048
#define DMODEL  1024
#define NHEADS  8
#define DHEAD   128
#define DLLM    4096
// reference scales by 1/sqrt(d_llm // H) = 1/sqrt(512)
#define SM_SCALE 0.044194173824159216f

// padded LDS stride (f16 elems) for the staged 64x64 B tile: 144B row stride
// keeps 16B alignment for v8h reads and spreads rows across all 64 banks.
#define BSTRIDE 72

__device__ inline v8f wmma_f16(v16h a, v16h b, v8f c) {
  return __builtin_amdgcn_wmma_f32_16x16x32_f16(
      /*neg_a=*/false, a, /*neg_b=*/false, b,
      /*c_mod=*/(short)0, c, /*reuse_a=*/false, /*reuse_b=*/false);
}

// A-fragment (16x32, 16-bit), contiguous K order (matches B frags):
// lane<16 : elems 0..15 = K 0..15
// lane>=16: elems 0..15 = K 16..31
template <typename T>
__device__ inline v16h load_afrag(const T* __restrict__ base, int koff) {
  v16h a;
#pragma unroll
  for (int i = 0; i < 8; ++i) a[i] = (_Float16)base[2 * koff + i];
#pragma unroll
  for (int i = 0; i < 8; ++i) a[8 + i] = (_Float16)base[2 * koff + 8 + i];
  return a;
}

// B-fragment (K=32 slice) from the staged [n][k] f16 tile; 2 x ds_load_b128.
__device__ inline v16h load_bfrag_lds(const _Float16* __restrict__ bl) {
  const v8h lo = *(const v8h*)bl;
  const v8h hi = *(const v8h*)(bl + 8);
  v16h bf;
#pragma unroll
  for (int i = 0; i < 8; ++i) { bf[i] = lo[i]; bf[8 + i] = hi[i]; }
  return bf;
}

// ---------------------------------------------------------------------------
// C = A[M,K](AT) * B[K,N](f32 -> f16) + bias[N]
//   TRANS == false : C[m*N + n]   (OT)
//   TRANS == true  : C[n*M + m]   (OT)   -- used for transposed V
// Block = 256 threads = 8 waves. Block tile 128(M) x 64(N); wave w owns rows
// [blockIdx.y*128 + w*16, +16). A 64-K-deep B tile (64x64 f32) is staged per
// step into LDS as [n][k] f16, giving 8 WMMAs per wave between barriers.
// ---------------------------------------------------------------------------
template <typename AT, typename OT, bool TRANS>
__global__ __launch_bounds__(256) void gemm_bias_kernel(
    const AT* __restrict__ A, const float* __restrict__ B,
    const float* __restrict__ bias, OT* __restrict__ C,
    int M, int N, int K) {
  __shared__ _Float16 lds_b[64 * BSTRIDE];

  const int lane  = threadIdx.x & 31;
  const int wave  = threadIdx.x >> 5;
  const int m0    = blockIdx.y * 128 + wave * 16;
  const int n0    = blockIdx.x * 64;
  const int mrow  = lane & 15;
  const int koff  = (lane >> 4) << 3;   // A frag: 0 or 8
  const int kboff = (lane >> 4) << 4;   // B frag: 0 or 16

  // cooperative B-stage mapping: 256 threads x 16 f32 = 64x64 tile
  const int bk = threadIdx.x >> 3;        // k row 0..31 (and +32)
  const int bn = (threadIdx.x & 7) * 8;   // n col 0,8,..,56

  v8f acc[4] = {};
  const AT* arow = A + (size_t)(m0 + mrow) * K;

  for (int k0 = 0; k0 < K; k0 += 64) {
    __syncthreads();
    {
      const float* g0 = B + (size_t)(k0 + bk) * N + n0 + bn;
      const float* g1 = g0 + (size_t)32 * N;
      const float4 x0 = *(const float4*)g0;
      const float4 x1 = *(const float4*)(g0 + 4);
      const float4 y0 = *(const float4*)g1;
      const float4 y1 = *(const float4*)(g1 + 4);
      if (k0 + 64 < K) {  // pull next tile toward cache while we compute
        __builtin_prefetch(g0 + (size_t)64 * N, 0, 1);
        __builtin_prefetch(g1 + (size_t)64 * N, 0, 1);
      }
      _Float16* d0 = lds_b + bk;        // element [n][k] at n*BSTRIDE + k
      _Float16* d1 = lds_b + bk + 32;
      d0[(bn + 0) * BSTRIDE] = (_Float16)x0.x;
      d0[(bn + 1) * BSTRIDE] = (_Float16)x0.y;
      d0[(bn + 2) * BSTRIDE] = (_Float16)x0.z;
      d0[(bn + 3) * BSTRIDE] = (_Float16)x0.w;
      d0[(bn + 4) * BSTRIDE] = (_Float16)x1.x;
      d0[(bn + 5) * BSTRIDE] = (_Float16)x1.y;
      d0[(bn + 6) * BSTRIDE] = (_Float16)x1.z;
      d0[(bn + 7) * BSTRIDE] = (_Float16)x1.w;
      d1[(bn + 0) * BSTRIDE] = (_Float16)y0.x;
      d1[(bn + 1) * BSTRIDE] = (_Float16)y0.y;
      d1[(bn + 2) * BSTRIDE] = (_Float16)y0.z;
      d1[(bn + 3) * BSTRIDE] = (_Float16)y0.w;
      d1[(bn + 4) * BSTRIDE] = (_Float16)y1.x;
      d1[(bn + 5) * BSTRIDE] = (_Float16)y1.y;
      d1[(bn + 6) * BSTRIDE] = (_Float16)y1.z;
      d1[(bn + 7) * BSTRIDE] = (_Float16)y1.w;
    }
    __syncthreads();

    const v16h af0 = load_afrag(arow + k0, koff);
    const v16h af1 = load_afrag(arow + k0 + 32, koff);
#pragma unroll
    for (int t = 0; t < 4; ++t) {
      const _Float16* bl = lds_b + (t * 16 + mrow) * BSTRIDE;
      acc[t] = wmma_f16(af0, load_bfrag_lds(bl + kboff), acc[t]);
      acc[t] = wmma_f16(af1, load_bfrag_lds(bl + 32 + kboff), acc[t]);
    }
  }

  const int rbase = m0 + koff;  // C layout: VGPR r -> row r (lanes<16) / 8+r
#pragma unroll
  for (int t = 0; t < 4; ++t) {
    const int col = n0 + t * 16 + mrow;
    const float bv = bias[col];
#pragma unroll
    for (int r = 0; r < 8; ++r) {
      const float val = acc[t][r] + bv;
      if (TRANS)
        C[(size_t)col * M + rbase + r] = (OT)val;
      else
        C[(size_t)(rbase + r) * N + col] = (OT)val;
    }
  }
}

// ---------------------------------------------------------------------------
// Attention: one block (8 waves) per (b, h, 16-row q tile).
//   scores[16,2048] = q_tile[16,128] * k_h[2048,128]^T   (WMMA, f32 -> LDS)
//   softmax(SM_SCALE * scores) rows in LDS
//   out[16,128]     = P[16,2048] * v_h[2048,128]         (WMMA, V from vt)
// vt is stored [h*128+e][s] so P.V B-fragments are single 32B loads.
// ---------------------------------------------------------------------------
__global__ __launch_bounds__(256) void attn_kernel(
    const _Float16* __restrict__ q, const _Float16* __restrict__ k,
    const _Float16* __restrict__ vt, _Float16* __restrict__ o) {
  extern __shared__ float smem[];  // 16 * 2048 f32 scores = 128 KB

  const int lane  = threadIdx.x & 31;
  const int wave  = threadIdx.x >> 5;
  const int lt    = blockIdx.x & 63;
  const int h     = (blockIdx.x >> 6) & 7;
  const int b     = blockIdx.x >> 9;
  const int l0    = lt * 16;
  const int mrow  = lane & 15;
  const int koff  = (lane >> 4) << 3;
  const int kboff = (lane >> 4) << 4;

  // q fragments: 16 rows, full E=128 (4 x K=32)
  const _Float16* qrow =
      q + (size_t)(b * LQ + l0 + mrow) * DMODEL + h * DHEAD;
  v16h qf[4];
#pragma unroll
  for (int t = 0; t < 4; ++t) qf[t] = load_afrag(qrow + t * 32, koff);

  // ---- scores: each wave covers 16 of the 128 column tiles -------------
  for (int ct = wave; ct < VS / 16; ct += 8) {
    const int s0 = ct * 16;
    v8f acc = {};
    const _Float16* kp =
        k + (size_t)(s0 + mrow) * DMODEL + h * DHEAD + kboff;
#pragma unroll
    for (int t = 0; t < 4; ++t) {
      const v16h bf = *(const v16h*)(kp + t * 32);
      acc = wmma_f16(qf[t], bf, acc);
    }
#pragma unroll
    for (int r = 0; r < 8; ++r)
      smem[(size_t)(r + koff) * VS + s0 + mrow] = acc[r];
  }
  __syncthreads();

  // ---- softmax: 256 threads = 16 rows x 16 lanes -----------------------
  {
    const int row = threadIdx.x >> 4;
    const int sub = threadIdx.x & 15;
    float* srow = smem + (size_t)row * VS;
    float m = -3.4e38f;
#pragma unroll 4
    for (int i = sub; i < VS; i += 16) m = fmaxf(m, srow[i] * SM_SCALE);
#pragma unroll
    for (int off = 8; off; off >>= 1) m = fmaxf(m, __shfl_xor(m, off, 16));
    float s = 0.0f;
#pragma unroll 4
    for (int i = sub; i < VS; i += 16) {
      const float p = __expf(srow[i] * SM_SCALE - m);
      srow[i] = p;
      s += p;
    }
#pragma unroll
    for (int off = 8; off; off >>= 1) s += __shfl_xor(s, off, 16);
    const float inv = 1.0f / s;
#pragma unroll 4
    for (int i = sub; i < VS; i += 16) srow[i] *= inv;
  }
  __syncthreads();

  // ---- out = P * V: wave w owns output cols [w*16, w*16+16) ------------
  const int n0 = wave * 16;
  v8f acc = {};
  const _Float16* vrow = vt + (size_t)(h * DHEAD + n0 + mrow) * VS;
  for (int k0 = 0; k0 < VS; k0 += 32) {
    const float* arow = smem + (size_t)mrow * VS + k0;
    v16h af;
#pragma unroll
    for (int i = 0; i < 8; ++i) af[i] = (_Float16)arow[2 * koff + i];
#pragma unroll
    for (int i = 0; i < 8; ++i) af[8 + i] = (_Float16)arow[2 * koff + 8 + i];
    const v16h bf = *(const v16h*)(vrow + k0 + kboff);  // contiguous 32B
    acc = wmma_f16(af, bf, acc);
  }
  const int rbase = l0 + koff;
#pragma unroll
  for (int r = 0; r < 8; ++r)
    o[(size_t)(b * LQ + rbase + r) * DMODEL + h * DHEAD + n0 + mrow] =
        (_Float16)acc[r];
}

// ---------------------------------------------------------------------------
extern "C" void kernel_launch(void* const* d_in, const int* in_sizes, int n_in,
                              void* d_out, int out_size, void* d_ws,
                              size_t ws_size, hipStream_t stream) {
  (void)in_sizes; (void)n_in; (void)out_size; (void)ws_size;
  const float* Q    = (const float*)d_in[0];
  const float* Ksrc = (const float*)d_in[1];
  const float* Vsrc = (const float*)d_in[2];
  const float* Wq   = (const float*)d_in[3];
  const float* bq   = (const float*)d_in[4];
  const float* Wk   = (const float*)d_in[5];
  const float* bk   = (const float*)d_in[6];
  const float* Wv   = (const float*)d_in[7];
  const float* bv   = (const float*)d_in[8];
  const float* Wo   = (const float*)d_in[9];
  const float* bo   = (const float*)d_in[10];
  float* out = (float*)d_out;

  // workspace: f16 q / k / vT / attn_out (40 MB total)
  _Float16* qh = (_Float16*)d_ws;
  _Float16* kh = qh + (size_t)BM * LQ * DMODEL;
  _Float16* vt = kh + (size_t)VS * DMODEL;         // [h*128+e][s]
  _Float16* ah = vt + (size_t)VS * DMODEL;

  const dim3 blk(256);

  gemm_bias_kernel<float, _Float16, false>
      <<<dim3(DMODEL / 64, (BM * LQ) / 128), blk, 0, stream>>>(
          Q, Wq, bq, qh, BM * LQ, DMODEL, DMODEL);
  gemm_bias_kernel<float, _Float16, false>
      <<<dim3(DMODEL / 64, VS / 128), blk, 0, stream>>>(
          Ksrc, Wk, bk, kh, VS, DMODEL, DLLM);
  gemm_bias_kernel<float, _Float16, true>
      <<<dim3(DMODEL / 64, VS / 128), blk, 0, stream>>>(
          Vsrc, Wv, bv, vt, VS, DMODEL, DLLM);

  attn_kernel<<<dim3(BM * NHEADS * (LQ / 16)), blk, 16 * VS * sizeof(float),
                stream>>>(qh, kh, vt, ah);

  gemm_bias_kernel<_Float16, float, false>
      <<<dim3(DLLM / 64, (BM * LQ) / 128), blk, 0, stream>>>(
          ah, Wo, bo, out, BM * LQ, DLLM, DMODEL);
}
